// ConvGeodesic_20401094656384
// MI455X (gfx1250) — matrix operand order repaired and checked by reading, not compiled, not run on hardware
//
#include <hip/hip_runtime.h>

// ---------------------------------------------------------------------------
// ConvGeodesic: B=1, m=100000, T=8, R=5, n=64, O=64, K=2
// out[t,mv,o] = relu( sum_{r,nn} interp[mv,t,r,nn] * Wsum[r,o,nn] + bias_term[o] )
// interp[mv,t,r,nn] = sum_c w[mv,t,r,c] * signal[idx[mv,t,r,c], nn]
// Implemented as an M=(T*m), K=320, N=64 GEMM on v_wmma_f32_16x16x32_f16.
// ---------------------------------------------------------------------------

typedef __attribute__((ext_vector_type(4)))  unsigned int u32x4;
typedef __attribute__((ext_vector_type(8)))  unsigned int u32x8;
typedef __attribute__((ext_vector_type(16))) _Float16     v16h;
typedef __attribute__((ext_vector_type(8)))  float        v8f;

#define T_DIM   8
#define R_DIM   5
#define N_DIM   64
#define O_DIM   64
#define KK_DIM  2
#define MTILE   64
#define KTOT    (R_DIM * N_DIM)       // 320 (GEMM K)
#define NKT     (KTOT / 32)           // 10 K-tiles of 32
#define NNT     (O_DIM / 16)          // 4 N-tiles of 16
#define A_STRIDE 328                  // 320 + 8 pad halves -> conflict-free b128 reads
#define A_BYTES  (MTILE * A_STRIDE * 2)   // 41984
#define B_WORDS  (NKT * NNT * 256)        // 10240 u32 (40 KB of B fragments)
#define WS_BIAS_OFF B_WORDS               // bias_term floats start here (u32 index)

static __device__ __forceinline__ unsigned int f2h_bits(float x) {
    _Float16 h = (_Float16)x;
    return (unsigned int)__builtin_bit_cast(unsigned short, h);
}

// ---------------------------------------------------------------------------
// Prep: Wsum = sum over (t,k) of W, converted to f16, written directly in the
// WMMA B-fragment layout (B is 32(K)x16(N); VGPR j of lane L holds K pair
// 16*(j/4)+8*(L/16)+2*(j%4), +1 (lanes 0-15: K 0-7,16-23; 16-31: K 8-15,24-31); N = L%16).
// ws u32 index = frag*256 + lane*8 + j, frag = kt*4 + nt.
// Also bias_term[o] = (K*R) * (bias[0,o] + bias[1,o]).
// ---------------------------------------------------------------------------
__global__ void conv_geo_prep(const float* __restrict__ W,
                              const float* __restrict__ bias,
                              unsigned int* __restrict__ ws) {
    int j = blockIdx.x * blockDim.x + threadIdx.x;
    if (j < B_WORDS) {
        int frag = j >> 8;
        int li   = j & 255;
        int lane = li >> 3;
        int v    = li & 7;
        int kt   = frag >> 2;
        int nt   = frag & 3;
        int col  = nt * 16 + (lane & 15);                 // O index
        int k0   = kt * 32 + ((v >> 2) << 4) + ((lane >> 4) << 3) + 2 * (v & 3);  // GEMM-K of low half
        int r0 = k0 >> 6,        n0 = k0 & 63;
        int r1 = (k0 + 1) >> 6,  n1 = (k0 + 1) & 63;
        float acc0 = 0.f, acc1 = 0.f;
        #pragma unroll
        for (int t = 0; t < T_DIM; ++t) {
            #pragma unroll
            for (int k = 0; k < KK_DIM; ++k) {
                const float* base = W + (((size_t)(t * KK_DIM + k) * R_DIM) * O_DIM) * N_DIM;
                acc0 += base[((size_t)r0 * O_DIM + col) * N_DIM + n0];
                acc1 += base[((size_t)r1 * O_DIM + col) * N_DIM + n1];
            }
        }
        ws[j] = f2h_bits(acc0) | (f2h_bits(acc1) << 16);
    } else if (j < B_WORDS + O_DIM) {
        int o = j - B_WORDS;
        float bt = (float)(KK_DIM * R_DIM) * (bias[o] + bias[O_DIM + o]);
        ((float*)ws)[WS_BIAS_OFF + o] = bt;
    }
}

// ---------------------------------------------------------------------------
// Main: one workgroup = one (t, 64-vertex tile).
//   Phase 1: gather + barycentric interp -> 64x320 f16 A-tile in LDS.
//   Phase 2: 8 waves x (16x32 C-tile), 10 wmma_f32_16x16x32_f16 steps each.
// ---------------------------------------------------------------------------
__global__ void __launch_bounds__(256)
conv_geo_main(const float* __restrict__ signal,
              const float* __restrict__ bary,
              const unsigned int* __restrict__ wsB,
              float* __restrict__ out, int m) {
    extern __shared__ char smem[];
    unsigned short* Ab = (unsigned short*)smem;                 // A tile, f16 bits
    unsigned int*   Bb = (unsigned int*)(smem + A_BYTES);       // B fragments

    const int tid = threadIdx.x;
    const int t   = blockIdx.y;
    const int m0  = blockIdx.x * MTILE;

    // ---- stage B fragments into LDS (L2-resident source, read once per WG)
    #pragma unroll 4
    for (int i = tid; i < B_WORDS; i += 256)
        Bb[i] = wsB[i];

    // ---- phase 1: 64 rows x 5 radial bins x 16 float4-chunks = 5120 tasks
    #pragma unroll 2
    for (int it = 0; it < 20; ++it) {
        int task = it * 256 + tid;
        int row  = task / 80;
        int rem  = task % 80;
        int r    = rem >> 4;
        int ch   = rem & 15;
        int mv   = m0 + row;
        float ax = 0.f, ay = 0.f, az = 0.f, aw = 0.f;
        if (mv < m) {
            const float* bb = bary + (((size_t)mv * T_DIM + t) * R_DIM + r) * 6;
            #pragma unroll
            for (int c = 0; c < 3; ++c) {
                int   ix = (int)bb[2 * c];
                float w  = bb[2 * c + 1];
                const float4 s = *(const float4*)(signal + (size_t)ix * N_DIM + ch * 4);
                ax = fmaf(w, s.x, ax);
                ay = fmaf(w, s.y, ay);
                az = fmaf(w, s.z, az);
                aw = fmaf(w, s.w, aw);
            }
        }
        unsigned int p0 = f2h_bits(ax) | (f2h_bits(ay) << 16);
        unsigned int p1 = f2h_bits(az) | (f2h_bits(aw) << 16);
        *(uint2*)(&Ab[row * A_STRIDE + (r * 64 + ch * 4)]) = make_uint2(p0, p1);
    }
    __syncthreads();

    // ---- phase 2: wave w owns rows r0..r0+15, cols c0..c0+31
    const int wave  = tid >> 5;
    const int lane  = tid & 31;
    const int r0    = (wave & 3) * 16;
    const int c0    = (wave >> 2) * 32;
    const int lhalf = lane >> 4;
    const int l16   = lane & 15;

    v8f acc0 = {};
    v8f acc1 = {};

    const unsigned short* aptr = Ab + (r0 + l16) * A_STRIDE;
    const int nt0 = c0 >> 4;

    #pragma unroll
    for (int kt = 0; kt < NKT; ++kt) {
        // A fragment: 16-bit A 16x32 layout — lanes 0-15: K 0-7 & 16-23,
        // lanes 16-31: K 8-15 & 24-31 (each run contiguous -> 2x ds_load_b128)
        int kb = kt * 32 + lhalf * 8;
        u32x4 alo = *(const u32x4*)(aptr + kb);
        u32x4 ahi = *(const u32x4*)(aptr + kb + 16);
        u32x8 au;
        au[0] = alo[0]; au[1] = alo[1]; au[2] = alo[2]; au[3] = alo[3];
        au[4] = ahi[0]; au[5] = ahi[1]; au[6] = ahi[2]; au[7] = ahi[3];
        v16h a = __builtin_bit_cast(v16h, au);

        // B fragments (pre-laid-out): 8 consecutive u32 per lane -> 2x b128
        const unsigned int* bp0 = Bb + (kt * NNT + nt0) * 256 + lane * 8;
        u32x4 b0lo = *(const u32x4*)bp0;
        u32x4 b0hi = *(const u32x4*)(bp0 + 4);
        u32x8 bu0;
        bu0[0] = b0lo[0]; bu0[1] = b0lo[1]; bu0[2] = b0lo[2]; bu0[3] = b0lo[3];
        bu0[4] = b0hi[0]; bu0[5] = b0hi[1]; bu0[6] = b0hi[2]; bu0[7] = b0hi[3];
        v16h b0 = __builtin_bit_cast(v16h, bu0);

        const unsigned int* bp1 = bp0 + 256;
        u32x4 b1lo = *(const u32x4*)bp1;
        u32x4 b1hi = *(const u32x4*)(bp1 + 4);
        u32x8 bu1;
        bu1[0] = b1lo[0]; bu1[1] = b1lo[1]; bu1[2] = b1lo[2]; bu1[3] = b1lo[3];
        bu1[4] = b1hi[0]; bu1[5] = b1hi[1]; bu1[6] = b1hi[2]; bu1[7] = b1hi[3];
        v16h b1 = __builtin_bit_cast(v16h, bu1);

        acc0 = __builtin_amdgcn_wmma_f32_16x16x32_f16(
            false, a, false, b0, (short)0, acc0, false, false);
        acc1 = __builtin_amdgcn_wmma_f32_16x16x32_f16(
            false, a, false, b1, (short)0, acc1, false, false);
    }

    // ---- epilogue: bias + relu + coalesced store (C layout: lane<16 -> M=v,
    // lane>=16 -> M=v+8; N = lane%16)
    const float* biasT = (const float*)wsB + WS_BIAS_OFF;
    const float  bia0  = biasT[c0 + l16];
    const float  bia1  = biasT[c0 + 16 + l16];

    #pragma unroll
    for (int v = 0; v < 8; ++v) {
        int row = r0 + v + lhalf * 8;
        int mv  = m0 + row;
        if (mv < m) {
            size_t o = ((size_t)t * (size_t)m + (size_t)mv) * O_DIM;
            float x0 = acc0[v] + bia0;
            float x1 = acc1[v] + bia1;
            out[o + c0 + l16]      = x0 > 0.f ? x0 : 0.f;
            out[o + c0 + 16 + l16] = x1 > 0.f ? x1 : 0.f;
        }
    }
}

// ---------------------------------------------------------------------------
extern "C" void kernel_launch(void* const* d_in, const int* in_sizes, int n_in,
                              void* d_out, int out_size, void* d_ws, size_t ws_size,
                              hipStream_t stream) {
    (void)n_in; (void)out_size; (void)ws_size;
    const float* signal = (const float*)d_in[0];
    const float* bary   = (const float*)d_in[1];
    const float* W      = (const float*)d_in[2];
    const float* bias   = (const float*)d_in[3];
    float* out          = (float*)d_out;
    unsigned int* ws    = (unsigned int*)d_ws;

    const int m = in_sizes[0] / N_DIM;   // B=1 -> m from signal size

    // 1) reduce W into f16 B-fragments + bias_term (needs ~41 KB of d_ws)
    int prepThreads = B_WORDS + O_DIM;
    conv_geo_prep<<<(prepThreads + 255) / 256, 256, 0, stream>>>(W, bias, ws);

    // 2) gather-interp + WMMA GEMM
    dim3 grid((m + MTILE - 1) / MTILE, T_DIM);
    size_t shmem = (size_t)A_BYTES + (size_t)B_WORDS * 4;   // 82944 B dynamic LDS
    conv_geo_main<<<grid, 256, shmem, stream>>>(signal, bary, ws, out, m);
}
